// NeighborAttention_27453430956543
// MI455X (gfx1250) — matrix operand order repaired and checked by reading, not compiled, not run on hardware
//
#include <hip/hip_runtime.h>

typedef __attribute__((ext_vector_type(16))) __bf16 v16bf;
typedef __attribute__((ext_vector_type(8)))  __bf16 v8bf;
typedef __attribute__((ext_vector_type(8)))  float  v8f;

// ---- fragment table (order must match k_pack decode) ----
// frag id within a layer = ktile*NT + ntile ; each fragment = 32 lanes * 32B = 1024B
#define FRAG_B0   0   // b0_w : K=192 (6 kt) x N=64 (4 nt) -> 24
#define FRAG_B1   24  // b1_w : K=64  (2 kt) x N=64 (4 nt) -> 8
#define FRAG_B2   32  // b2_w : K=64  (2 kt) x N=4 pad16 (1 nt) -> 2
#define FRAG_WV0  34  // wv0_w: K=128 (4 kt) x N=64 (4 nt) -> 16
#define FRAG_WV1  50  // wv1_w: 8
#define FRAG_WV2  58  // wv2_w: 8
#define FRAG_WO   66  // wo_w : 8
#define NFRAG     74

// ---------------- helpers ----------------
__device__ inline unsigned short f2bf(float x) {
  union { __bf16 h; unsigned short u; } c; c.h = (__bf16)x; return c.u;
}
__device__ inline float bf2f(unsigned short v) {
  union { __bf16 h; unsigned short u; } c; c.u = v; return (float)c.h;
}
__device__ inline unsigned int packpair(float a, float b) {
  return (unsigned int)f2bf(a) | ((unsigned int)f2bf(b) << 16);
}
__device__ inline uint4 pack8(const float4& a, const float4& b) {
  uint4 r;
  r.x = packpair(a.x, a.y); r.y = packpair(a.z, a.w);
  r.z = packpair(b.x, b.y); r.w = packpair(b.z, b.w);
  return r;
}
// fast reciprocal (v_rcp_f32): plenty for bf16-bound results
__device__ inline float frcp(float x) { return __builtin_amdgcn_rcpf(x); }
__device__ inline float silu(float x) { return x * frcp(1.f + __expf(-x)); }

union AFrag { v16bf v; v8bf h[2]; };
union BFrag { v16bf v; uint4 q[2]; };

// Load A fragments (16x32 bf16, row-major in LDS) per ISA 16-bit A layout:
// lane<16 : M=lane,   K = kt*32 + [0..7]  and [16..23]
// lane>=16: M=lane-16,K = kt*32 + [8..15] and [24..31]
template<int KT>
__device__ inline void load_afrags(AFrag (&af)[KT], const unsigned short* base,
                                   int stride, int lane) {
  const int row  = lane & 15;
  const int hsel = lane >> 4;
  const unsigned short* rp = base + row * stride;
#pragma unroll
  for (int kt = 0; kt < KT; ++kt) {
    const unsigned short* p = rp + kt * 32 + hsel * 8;
    af[kt].h[0] = *(const v8bf*)(p);
    af[kt].h[1] = *(const v8bf*)(p + 16);
  }
}

// One 16x16 output tile: acc = A(16x32*KT) x B + bias-broadcast (B from LDS or global)
template<int KT, int NT>
__device__ inline v8f mm_tile(const AFrag (&af)[KT], const uint4* wf,
                              int fragBase, int nt, float bv, int lane) {
  v8f acc = { bv, bv, bv, bv, bv, bv, bv, bv };
#pragma unroll
  for (int kt = 0; kt < KT; ++kt) {
    BFrag b;
    const uint4* p = wf + (size_t)(fragBase + kt * NT + nt) * 64 + lane * 2;
    b.q[0] = p[0];
    b.q[1] = p[1];
    acc = __builtin_amdgcn_wmma_f32_16x16x32_bf16(false, af[kt].v, false, b.v,
                                                  (short)0, acc, false, false);
  }
  return acc;
}

// Store f32 D tile (VGPR r: M=r or r+8 per half, N=lane&15) to bf16 LDS row-major
__device__ inline void store_act(const v8f& acc, unsigned short* out, int stride,
                                 int nt, int lane, bool do_silu) {
  const int col  = lane & 15;
  const int hsel = lane >> 4;
#pragma unroll
  for (int r = 0; r < 8; ++r) {
    float x = acc[r];
    if (do_silu) x = silu(x);
    out[(r + hsel * 8) * stride + nt * 16 + col] = f2bf(x);
  }
}

// ---------------- kernel 1: pack weights into B-fragment layout ----------------
// B (32x16 bf16) per ISA: lane<16 holds N=lane, K=[0..7],[16..23] (2 per dword, even K low);
// lane>=16 holds N=lane-16, K=[8..15],[24..31].
__global__ __launch_bounds__(256) void k_pack(
    const float* __restrict__ b0, const float* __restrict__ b1,
    const float* __restrict__ b2, const float* __restrict__ wv0,
    const float* __restrict__ wv1, const float* __restrict__ wv2,
    const float* __restrict__ wo, unsigned int* __restrict__ wout) {
  int t = blockIdx.x * 256 + threadIdx.x;
  if (t >= NFRAG * 32) return;
  int f = t >> 5, lane = t & 31;
  const float* W; int Nout, NT, base;
  if      (f < 24) { W = b0;  base = FRAG_B0;  NT = 4; Nout = 64; }
  else if (f < 32) { W = b1;  base = FRAG_B1;  NT = 4; Nout = 64; }
  else if (f < 34) { W = b2;  base = FRAG_B2;  NT = 1; Nout = 4;  }
  else if (f < 50) { W = wv0; base = FRAG_WV0; NT = 4; Nout = 64; }
  else if (f < 58) { W = wv1; base = FRAG_WV1; NT = 4; Nout = 64; }
  else if (f < 66) { W = wv2; base = FRAG_WV2; NT = 4; Nout = 64; }
  else             { W = wo;  base = FRAG_WO;  NT = 4; Nout = 64; }
  int fl = f - base, kt = fl / NT, nt = fl % NT;
  int n = nt * 16 + (lane & 15);
#pragma unroll
  for (int j = 0; j < 8; ++j) {
    int k = kt * 32 + 2 * j + ((lane & 16) ? 8 : 0) + ((j >= 4) ? 8 : 0);
    float w0 = (n < Nout) ? W[(size_t)k * Nout + n]       : 0.f;
    float w1 = (n < Nout) ? W[(size_t)(k + 1) * Nout + n] : 0.f;
    wout[(size_t)t * 8 + j] = packpair(w0, w1);
  }
}

// ---------------- kernel 2: zero degree counters ----------------
__global__ __launch_bounds__(256) void k_zero(int* __restrict__ deg, int N) {
  int t = blockIdx.x * 256 + threadIdx.x;
  if (t < N) deg[t] = 0;
}

// ---------------- kernel 3: edge MLPs (bias logits + values) ----------------
// 8 waves/block, 2 tiles of 16 edges per wave => 256 edges/block.
// All weight fragments staged once into LDS per block.
__global__ __launch_bounds__(256) void k_edge(
    const float* __restrict__ hV, const float* __restrict__ hE,
    const int* __restrict__ cid, const uint4* __restrict__ wf,
    const float* __restrict__ b0b, const float* __restrict__ b1b,
    const float* __restrict__ b2b, const float* __restrict__ wv0b,
    const float* __restrict__ wv1b, const float* __restrict__ wv2b,
    float* __restrict__ logits, unsigned short* __restrict__ vbuf,
    int* __restrict__ deg, int E) {
  __shared__ __attribute__((aligned(16))) uint4 wlds[NFRAG * 64];          // 75776 B
  __shared__ __attribute__((aligned(16))) unsigned short xbuf[8][16][192]; // 49152 B
  __shared__ __attribute__((aligned(16))) unsigned short abuf[8][2][16][64]; // 32768 B
  __shared__ int cbuf[8][16];                                              // 512 B
  // stage all weight fragments into LDS
  for (int i = threadIdx.x; i < NFRAG * 64; i += 256) wlds[i] = wf[i];
  __syncthreads();

  const int wave = threadIdx.x >> 5;
  const int lane = threadIdx.x & 31;
  const int row  = lane & 15;
  const int hsel = lane >> 4;
  const int col  = row;

#pragma clang loop unroll(disable)
  for (int it = 0; it < 2; ++it) {
    const int e0 = (blockIdx.x * 16 + wave * 2 + it) * 16;
    const int er = min(e0 + row, E - 1);
    const int c  = cid[er];
    if (hsel == 0) cbuf[wave][row] = c;

    // stage bf16 X = [h_V[center] ; h_E] into LDS (16 x 192)
    {
      const float* src = hV + (size_t)c * 64 + hsel * 32;
      unsigned short* dst = &xbuf[wave][row][hsel * 32];
#pragma unroll
      for (int k = 0; k < 32; k += 8) {
        float4 f0 = *(const float4*)(src + k);
        float4 f1 = *(const float4*)(src + k + 4);
        *(uint4*)(dst + k) = pack8(f0, f1);
      }
    }
    {
      const float* src = hE + (size_t)er * 128 + hsel * 64;
      unsigned short* dst = &xbuf[wave][row][64 + hsel * 64];
#pragma unroll
      for (int k = 0; k < 64; k += 8) {
        float4 f0 = *(const float4*)(src + k);
        float4 f1 = *(const float4*)(src + k + 4);
        *(uint4*)(dst + k) = pack8(f0, f1);
      }
    }

    // ---- bias MLP: x(192) -> 64 -> 64 -> 4 logits ----
    {
      AFrag af[6];
      load_afrags<6>(af, &xbuf[wave][0][0], 192, lane);
#pragma unroll
      for (int nt = 0; nt < 4; ++nt) {
        v8f acc = mm_tile<6, 4>(af, wlds, FRAG_B0, nt, b0b[nt * 16 + col], lane);
        store_act(acc, &abuf[wave][0][0][0], 64, nt, lane, true);
      }
    }
    {
      AFrag af[2];
      load_afrags<2>(af, &abuf[wave][0][0][0], 64, lane);
#pragma unroll
      for (int nt = 0; nt < 4; ++nt) {
        v8f acc = mm_tile<2, 4>(af, wlds, FRAG_B1, nt, b1b[nt * 16 + col], lane);
        store_act(acc, &abuf[wave][1][0][0], 64, nt, lane, true);
      }
    }
    {
      AFrag af[2];
      load_afrags<2>(af, &abuf[wave][1][0][0], 64, lane);
      float bv = (col < 4) ? b2b[col] : 0.f;
      v8f acc = mm_tile<2, 1>(af, wlds, FRAG_B2, 0, bv, lane);
      if (col < 4) {
#pragma unroll
        for (int r = 0; r < 8; ++r) {
          int M = r + hsel * 8;
          int e = e0 + M;
          if (e < E) {
            logits[(size_t)e * 4 + col] = acc[r] * 0.25f;  // 1/sqrt(HEAD_D)
            if (col == 0) atomicAdd(&deg[cbuf[wave][M]], 1);
          }
        }
      }
    }

    // ---- value MLP: h_E(128) -> 64 -> 64 -> 64 ----
    {
      AFrag af[4];
      load_afrags<4>(af, &xbuf[wave][0][64], 192, lane);
#pragma unroll
      for (int nt = 0; nt < 4; ++nt) {
        v8f acc = mm_tile<4, 4>(af, wlds, FRAG_WV0, nt, wv0b[nt * 16 + col], lane);
        store_act(acc, &abuf[wave][0][0][0], 64, nt, lane, true);
      }
    }
    {
      AFrag af[2];
      load_afrags<2>(af, &abuf[wave][0][0][0], 64, lane);
#pragma unroll
      for (int nt = 0; nt < 4; ++nt) {
        v8f acc = mm_tile<2, 4>(af, wlds, FRAG_WV1, nt, wv1b[nt * 16 + col], lane);
        store_act(acc, &abuf[wave][1][0][0], 64, nt, lane, true);
      }
    }
    {
      AFrag af[2];
      load_afrags<2>(af, &abuf[wave][1][0][0], 64, lane);
#pragma unroll
      for (int nt = 0; nt < 4; ++nt) {
        v8f acc = mm_tile<2, 4>(af, wlds, FRAG_WV2, nt, wv2b[nt * 16 + col], lane);
        store_act(acc, &abuf[wave][0][0][0], 64, nt, lane, false);  // no silu
      }
      // coalesced bf16 row copy LDS -> global (4x b128 per lane)
      int e = e0 + row;
      if (e < E) {
        const unsigned short* srow = &abuf[wave][0][row][hsel * 32];
        uint4* dstg = (uint4*)(vbuf + (size_t)e * 64 + hsel * 32);
#pragma unroll
        for (int k = 0; k < 32; k += 8) dstg[k / 8] = *(const uint4*)(srow + k);
      }
    }
  }
}

// ---------------- CSR build: scan kernels + fill ----------------
__global__ __launch_bounds__(256) void k_scan1(const int* __restrict__ deg,
                                               int* __restrict__ incl,
                                               int* __restrict__ bsum, int N) {
  __shared__ int sh[256];
  int g = blockIdx.x * 256 + threadIdx.x;
  sh[threadIdx.x] = (g < N) ? deg[g] : 0;
  __syncthreads();
#pragma unroll
  for (int off = 1; off < 256; off <<= 1) {
    int t = (threadIdx.x >= off) ? sh[threadIdx.x - off] : 0;
    __syncthreads();
    sh[threadIdx.x] += t;
    __syncthreads();
  }
  if (g < N) incl[g] = sh[threadIdx.x];
  if (threadIdx.x == 255) bsum[blockIdx.x] = sh[255];
}

__global__ void k_scan2(const int* __restrict__ bsum, int* __restrict__ bpre,
                        int nb) {
  if (blockIdx.x == 0 && threadIdx.x == 0) {
    int run = 0;
    for (int i = 0; i < nb; ++i) { bpre[i] = run; run += bsum[i]; }
  }
}

__global__ __launch_bounds__(256) void k_scan3(
    const int* __restrict__ deg, const int* __restrict__ incl,
    const int* __restrict__ bpre, int* __restrict__ start,
    int* __restrict__ cursor, int N) {
  int g = blockIdx.x * 256 + threadIdx.x;
  if (g < N) {
    int s = bpre[g >> 8] + incl[g] - deg[g];  // exclusive start
    start[g] = s;
    cursor[g] = s;
  }
}

__global__ __launch_bounds__(256) void k_fill(const int* __restrict__ cid,
                                              int* __restrict__ cursor,
                                              int* __restrict__ elist, int E) {
  int e = blockIdx.x * 256 + threadIdx.x;
  if (e < E) {
    int p = atomicAdd(&cursor[cid[e]], 1);
    elist[p] = e;
  }
}

// ---------------- kernel: gather-side softmax + aggregation ----------------
// thread (node, j): two passes over the node's edge list; no float atomics.
__global__ __launch_bounds__(256) void k_agg(
    const int* __restrict__ start, const int* __restrict__ deg,
    const int* __restrict__ elist, const float* __restrict__ logits,
    const unsigned short* __restrict__ vbuf, unsigned short* __restrict__ aggn,
    int N) {
  int t = blockIdx.x * 256 + threadIdx.x;
  int node = t >> 6, j = t & 63, h = j >> 4;
  if (node >= N) return;
  int s = start[node], d = deg[node];
  float m = -1e30f;
  for (int i = 0; i < d; ++i) {
    int e = elist[s + i];
    m = fmaxf(m, logits[(size_t)e * 4 + h]);
  }
  float acc = 0.f, ds = 0.f;
  for (int i = 0; i < d; ++i) {
    int e = elist[s + i];
    float ex = __expf(logits[(size_t)e * 4 + h] - m);
    acc += ex * bf2f(vbuf[(size_t)e * 64 + j]);
    ds += ex;
  }
  float r = (d > 0) ? acc * frcp(ds) : 0.f;  // softmax-weighted mean
  aggn[(size_t)node * 64 + j] = f2bf(r);
}

// ---------------- kernel: output projection (W_O) ----------------
__global__ __launch_bounds__(256) void k_node(
    const unsigned short* __restrict__ aggn, const uint4* __restrict__ wf,
    float* __restrict__ out, int N) {
  __shared__ __attribute__((aligned(16))) unsigned short nbuf[8][16][64];
  const int wave = threadIdx.x >> 5;
  const int lane = threadIdx.x & 31;
  const int row  = lane & 15;
  const int hsel = lane >> 4;
  const int col  = row;
  const int n0   = (blockIdx.x * 8 + wave) * 16;
  const int nr   = min(n0 + row, N - 1);
  const unsigned short* src = aggn + (size_t)nr * 64 + hsel * 32;
  unsigned short* dst = &nbuf[wave][row][hsel * 32];
#pragma unroll
  for (int k = 0; k < 32; k += 8)
    *(uint4*)(dst + k) = *(const uint4*)(src + k);
  AFrag af[2];
  load_afrags<2>(af, &nbuf[wave][0][0], 64, lane);
#pragma unroll
  for (int nt = 0; nt < 4; ++nt) {
    v8f acc = mm_tile<2, 4>(af, wf, FRAG_WO, nt, 0.f, lane);
#pragma unroll
    for (int r = 0; r < 8; ++r) {
      int M = r + hsel * 8;
      int n = n0 + M;
      if (n < N) out[(size_t)n * 64 + nt * 16 + col] = acc[r];
    }
  }
}

// ---------------- launcher ----------------
extern "C" void kernel_launch(void* const* d_in, const int* in_sizes, int n_in,
                              void* d_out, int out_size, void* d_ws, size_t ws_size,
                              hipStream_t stream) {
  const float* hV   = (const float*)d_in[0];
  const float* hE   = (const float*)d_in[1];
  const int*   cid  = (const int*)d_in[2];
  const float* wv0w = (const float*)d_in[3];  const float* wv0b = (const float*)d_in[4];
  const float* wv1w = (const float*)d_in[5];  const float* wv1b = (const float*)d_in[6];
  const float* wv2w = (const float*)d_in[7];  const float* wv2b = (const float*)d_in[8];
  const float* b0w  = (const float*)d_in[9];  const float* b0b  = (const float*)d_in[10];
  const float* b1w  = (const float*)d_in[11]; const float* b1b  = (const float*)d_in[12];
  const float* b2w  = (const float*)d_in[13]; const float* b2b  = (const float*)d_in[14];
  const float* wow  = (const float*)d_in[15];
  const int N = in_sizes[0] / 64;
  const int E = in_sizes[2];
  const int NB = (N + 255) / 256;

  char* ws = (char*)d_ws;
  auto aup = [](size_t x) { return (x + 255) & ~(size_t)255; };
  size_t off = aup((size_t)NFRAG * 1024);
  size_t off_logits = off; off += aup((size_t)E * 4 * sizeof(float));
  size_t off_v      = off; off += aup((size_t)E * 64 * sizeof(unsigned short));
  size_t off_deg    = off; off += aup((size_t)N * sizeof(int));
  size_t off_incl   = off; off += aup((size_t)N * sizeof(int));
  size_t off_bsum   = off; off += aup((size_t)NB * sizeof(int));
  size_t off_bpre   = off; off += aup((size_t)NB * sizeof(int));
  size_t off_start  = off; off += aup((size_t)N * sizeof(int));
  size_t off_cursor = off; off += aup((size_t)N * sizeof(int));
  size_t off_elist  = off; off += aup((size_t)E * sizeof(int));
  size_t off_aggn   = off; off += aup((size_t)N * 64 * sizeof(unsigned short));
  (void)ws_size; (void)n_in; (void)out_size;

  unsigned int*   wpack  = (unsigned int*)ws;
  float*          logit  = (float*)(ws + off_logits);
  unsigned short* vbuf   = (unsigned short*)(ws + off_v);
  int* deg    = (int*)(ws + off_deg);
  int* incl   = (int*)(ws + off_incl);
  int* bsum   = (int*)(ws + off_bsum);
  int* bpre   = (int*)(ws + off_bpre);
  int* start  = (int*)(ws + off_start);
  int* cursor = (int*)(ws + off_cursor);
  int* elist  = (int*)(ws + off_elist);
  unsigned short* aggn = (unsigned short*)(ws + off_aggn);

  k_pack<<<(NFRAG * 32 + 255) / 256, 256, 0, stream>>>(b0w, b1w, b2w, wv0w, wv1w,
                                                       wv2w, wow, wpack);
  k_zero<<<(N + 255) / 256, 256, 0, stream>>>(deg, N);
  k_edge<<<(E + 255) / 256, 256, 0, stream>>>(hV, hE, cid, (const uint4*)wpack,
                                              b0b, b1b, b2b, wv0b, wv1b, wv2b,
                                              logit, vbuf, deg, E);
  k_scan1<<<NB, 256, 0, stream>>>(deg, incl, bsum, N);
  k_scan2<<<1, 32, 0, stream>>>(bsum, bpre, NB);
  k_scan3<<<NB, 256, 0, stream>>>(deg, incl, bpre, start, cursor, N);
  k_fill<<<(E + 255) / 256, 256, 0, stream>>>(cid, cursor, elist, E);
  k_agg<<<(N * 64 + 255) / 256, 256, 0, stream>>>(start, deg, elist, logit, vbuf,
                                                  aggn, N);
  k_node<<<(N + 127) / 128, 256, 0, stream>>>(aggn, (const uint4*)wpack,
                                              (float*)d_out, N);
}
